// MambaImageClassifier_3410204033450
// MI455X (gfx1250) — matrix operand order repaired and checked by reading, not compiled, not run on hardware
//
#include <hip/hip_runtime.h>
#include <hip/hip_bf16.h>

// ---------------------------------------------------------------------------
// MambaImageClassifier forward for MI455X (gfx1250, wave32, WMMA).
// GEMMs: bf16 operands, f32 accumulate via v_wmma_f32_16x16x32_bf16.
// A-tiles staged global->LDS via global_load_async_to_lds_b128 (ASYNCcnt).
// ---------------------------------------------------------------------------

typedef __bf16 bf16x16 __attribute__((ext_vector_type(16)));
typedef float  f32x8   __attribute__((ext_vector_type(8)));
typedef int    v4i     __attribute__((ext_vector_type(4)));

#define BATCH  32
#define SEQ    256
#define MTOK   (BATCH * SEQ)   // 8192 tokens
#define DMODEL 768
#define DSTATE 16
#define DHID   1536
#define PDIM   588             // 3*14*14
#define NCLS   1000
#define NLAYER 8
#define IMGSZ  224
#define PATCH  14
#define HP     16              // IMG/P

#if defined(__gfx1250__) && __has_builtin(__builtin_amdgcn_global_load_async_to_lds_b128)
#define USE_ASYNC_LDS 1
typedef __attribute__((address_space(1))) v4i* gas_v4i_ptr;  // global (AS1)
typedef __attribute__((address_space(3))) v4i* las_v4i_ptr;  // LDS (AS3)
#else
#define USE_ASYNC_LDS 0
#endif

struct U32x8 { uint4 a, b; };

// load 2x16B from LDS and view as a bf16x16 fragment
__device__ __forceinline__ bf16x16 ld_frag(const __bf16* p0, const __bf16* p1) {
    U32x8 u;
    u.a = *(const uint4*)p0;
    u.b = *(const uint4*)p1;
    return __builtin_bit_cast(bf16x16, u);
}

// ---------------------------------------------------------------------------
// Generic tiled WMMA GEMM:  C[M,N] = A_bf16[M,K] * W_f32[N,K]^T  (+ epilogue)
//   MODE 0: outF = acc + bias (+ pos[(m%256)*N + n])                (f32 out)
//   MODE 1: outB = bf16(acc + bias)                                 (bf16 out)
//   MODE 2: outB = bf16(acc + bias + xn[m,n]*Dv[n])                 (ssm y)
//   MODE 3: outF[m,n] += acc + bias                                 (residual)
// Block = 256 threads (8 waves), tile 128x128, K-step 32.
// Waves arranged 4x2; each wave owns a 32x64 region = 2x4 tiles of 16x16.
// LDS layout: both tiles row-major [row][k], padded stride 40 bf16 (80 B),
// so every WMMA fragment is two contiguous ds_load_b128 per lane.
// ---------------------------------------------------------------------------
template<int MODE>
__global__ __launch_bounds__(256)
void gemm_wmma(const __hip_bfloat16* __restrict__ Ag, int lda,
               const float* __restrict__ Wg,
               const float* __restrict__ bias,
               float* __restrict__ outF,
               __hip_bfloat16* __restrict__ outB,
               const float* __restrict__ pos,
               const __hip_bfloat16* __restrict__ xnP,
               const float* __restrict__ Dv,
               int M, int N, int K)
{
    constexpr int AST = 40;   // LDS row stride (bf16 elems): 32 + 8 pad
    constexpr int WST = 40;
    __shared__ __align__(16) __bf16 lA[128 * AST];  // A tile  [m][k]
    __shared__ __align__(16) __bf16 lW[128 * WST];  // W tile  [n][k]

    const int tid  = threadIdx.x;
    const int lane = tid & 31;
    const int wid  = tid >> 5;
    const int wr   = wid >> 1;          // wave row 0..3
    const int wc   = wid & 1;           // wave col 0..1
    const int m0   = blockIdx.y * 128;
    const int n0   = blockIdx.x * 128;
    const int hl   = lane >> 4;         // lane half
    const int l16  = lane & 15;

    f32x8 acc[2][4] = {};

    const unsigned short* Au = (const unsigned short*)Ag;

    for (int k0 = 0; k0 < K; k0 += 32) {
        // ---- stage A tile: 128x32 bf16, raw copy (async when possible) ---
        {
            const int row = tid >> 1;
            const int cs  = (tid & 1) << 4;        // 0 or 16
            const int mg  = m0 + row;
            __bf16* dst = &lA[row * AST + cs];
            if (mg < M && (k0 + 32) <= K) {
                const unsigned short* src = Au + (size_t)mg * lda + k0 + cs;
#if USE_ASYNC_LDS
                // INST_OFFSET is added to both global and LDS addresses (ISA 10.x)
                __builtin_amdgcn_global_load_async_to_lds_b128(
                    (gas_v4i_ptr)src, (las_v4i_ptr)dst, 0, 0);
                __builtin_amdgcn_global_load_async_to_lds_b128(
                    (gas_v4i_ptr)src, (las_v4i_ptr)dst, 16, 0);
#else
                const uint2* p = (const uint2*)src;
                uint2* d = (uint2*)dst;
                d[0] = p[0]; d[1] = p[1]; d[2] = p[2]; d[3] = p[3];
#endif
                if (k0 + 32 < K)  // prefetch next K tile of this row
                    __builtin_prefetch((const void*)(src + 32), 0, 1);
            } else {
                for (int j = 0; j < 16; ++j) {
                    const int kg = k0 + cs + j;
                    unsigned short v = (mg < M && kg < K)
                                       ? Au[(size_t)mg * lda + kg] : (unsigned short)0;
                    ((unsigned short*)dst)[j] = v;
                }
            }
        }
        // ---- stage W tile: f32 -> bf16, LDS[n][k] (no transpose) ---------
        {
            const int n  = tid >> 1;
            const int ks = (tid & 1) << 4;
            const int ng = n0 + n;
            if (ng < N && (k0 + 32) <= K) {
                const float4* p = (const float4*)(Wg + (size_t)ng * K + k0 + ks);
                bf16x16 w;
#pragma unroll
                for (int q = 0; q < 4; ++q) {
                    float4 f = p[q];
                    w[4 * q + 0] = (__bf16)f.x;
                    w[4 * q + 1] = (__bf16)f.y;
                    w[4 * q + 2] = (__bf16)f.z;
                    w[4 * q + 3] = (__bf16)f.w;
                }
                U32x8 u = __builtin_bit_cast(U32x8, w);
                uint4* d = (uint4*)&lW[n * WST + ks];
                d[0] = u.a; d[1] = u.b;
                if (k0 + 32 < K)
                    __builtin_prefetch((const void*)(Wg + (size_t)ng * K + k0 + 32 + ks), 0, 1);
            } else {
                for (int j = 0; j < 16; ++j) {
                    const int kg = k0 + ks + j;
                    float f = (ng < N && kg < K) ? Wg[(size_t)ng * K + kg] : 0.0f;
                    lW[n * WST + ks + j] = (__bf16)f;
                }
            }
        }
#if USE_ASYNC_LDS
#if __has_builtin(__builtin_amdgcn_s_wait_asynccnt)
        __builtin_amdgcn_s_wait_asynccnt(0);
#else
        asm volatile("s_wait_asynccnt 0x0" ::: "memory");
#endif
#endif
        __syncthreads();

        // ---- fragments per ISA 7.12.2 (wave32, 16-bit A 16x32 / B 32x16) -
        // A: lane holds row m=l16, k = hl*8+{0..7} and 16+hl*8+{0..7}
        // B: lane holds col n=l16, k = hl*8+{0..7} and 16+hl*8+{0..7}
        bf16x16 af[2];
#pragma unroll
        for (int i = 0; i < 2; ++i) {
            const int row = wr * 32 + i * 16 + l16;
            const __bf16* base = &lA[row * AST + hl * 8];
            af[i] = ld_frag(base, base + 16);
        }
        bf16x16 bfm[4];
#pragma unroll
        for (int j = 0; j < 4; ++j) {
            const int col = wc * 64 + j * 16 + l16;
            const __bf16* base = &lW[col * WST + hl * 8];
            bfm[j] = ld_frag(base, base + 16);
        }
#pragma unroll
        for (int i = 0; i < 2; ++i)
#pragma unroll
            for (int j = 0; j < 4; ++j)
                acc[i][j] = __builtin_amdgcn_wmma_f32_16x16x32_bf16(
                    false, af[i], false, bfm[j], (short)0, acc[i][j], false, false);
        __syncthreads();
    }

    // ---- epilogue (C/D layout: vgpr r -> m = base + hl*8 + r, n = l16) ---
#pragma unroll
    for (int i = 0; i < 2; ++i) {
#pragma unroll
        for (int j = 0; j < 4; ++j) {
            const int mbase = m0 + wr * 32 + i * 16 + hl * 8;
            const int n     = n0 + wc * 64 + j * 16 + l16;
            if (n >= N) continue;
            const float bv = bias ? bias[n] : 0.0f;
#pragma unroll
            for (int r = 0; r < 8; ++r) {
                const int m = mbase + r;
                if (m >= M) continue;
                float v = acc[i][j][r] + bv;
                if (MODE == 0) {
                    if (pos) v += pos[(size_t)(m & (SEQ - 1)) * N + n];
                    outF[(size_t)m * N + n] = v;
                } else if (MODE == 1) {
                    outB[(size_t)m * N + n] = __float2bfloat16(v);
                } else if (MODE == 2) {
                    v += __bfloat162float(xnP[(size_t)m * N + n]) * Dv[n];
                    outB[(size_t)m * N + n] = __float2bfloat16(v);
                } else {
                    outF[(size_t)m * N + n] += v;
                }
            }
        }
    }
}

// ---------------------------------------------------------------------------
// Elementwise / sequence kernels
// ---------------------------------------------------------------------------
__device__ __forceinline__ float wave_sum(float v) {
    for (int m = 16; m > 0; m >>= 1) v += __shfl_xor(v, m, 32);
    return v;
}

// patches[t, p] in (c, kh, kw) order, tokens row-major over (h, w)
__global__ __launch_bounds__(256)
void patchify_kernel(const float* __restrict__ imgs, __hip_bfloat16* __restrict__ out)
{
    const size_t idx = (size_t)blockIdx.x * 256 + threadIdx.x;
    if (idx >= (size_t)MTOK * PDIM) return;
    const int t = (int)(idx / PDIM);
    const int p = (int)(idx % PDIM);
    const int b = t / SEQ, s = t % SEQ;
    const int hp = s / HP, wp = s % HP;
    const int c = p / (PATCH * PATCH);
    const int r = p % (PATCH * PATCH);
    const int kh = r / PATCH, kw = r % PATCH;
    const float v = imgs[(((size_t)b * 3 + c) * IMGSZ + hp * PATCH + kh) * IMGSZ
                         + wp * PATCH + kw];
    out[idx] = __float2bfloat16(v);
}

// one wave per token; 768 = 32 lanes x 24
__global__ __launch_bounds__(256)
void ln_kernel(const float* __restrict__ x, const float* __restrict__ g,
               const float* __restrict__ b, __hip_bfloat16* __restrict__ out)
{
    const int lane = threadIdx.x & 31, wid = threadIdx.x >> 5;
    const int t = blockIdx.x * 8 + wid;
    if (t >= MTOK) return;
    const float* xr = x + (size_t)t * DMODEL;
    float v[24];
    float s = 0.0f;
#pragma unroll
    for (int i = 0; i < 24; ++i) { v[i] = xr[lane + 32 * i]; s += v[i]; }
    const float mean = wave_sum(s) * (1.0f / DMODEL);
    float q = 0.0f;
#pragma unroll
    for (int i = 0; i < 24; ++i) { const float d = v[i] - mean; q += d * d; }
    const float inv = rsqrtf(wave_sum(q) * (1.0f / DMODEL) + 1e-5f);
#pragma unroll
    for (int i = 0; i < 24; ++i) {
        const int n = lane + 32 * i;
        out[(size_t)t * DMODEL + n] =
            __float2bfloat16((v[i] - mean) * inv * g[n] + b[n]);
    }
}

// delta[t] = sigmoid(dot(xn[t], tw) + tb) * 0.99 + 0.01   (one wave per token)
__global__ __launch_bounds__(256)
void delta_kernel(const __hip_bfloat16* __restrict__ xn,
                  const float* __restrict__ tw, const float* __restrict__ tb,
                  float* __restrict__ delta)
{
    const int lane = threadIdx.x & 31, wid = threadIdx.x >> 5;
    const int t = blockIdx.x * 8 + wid;
    if (t >= MTOK) return;
    const __hip_bfloat16* xr = xn + (size_t)t * DMODEL;
    float s = 0.0f;
#pragma unroll
    for (int i = 0; i < 24; ++i) {
        const int n = lane + 32 * i;
        s += __bfloat162float(xr[n]) * tw[n];
    }
    s = wave_sum(s);
    if (lane == 0) {
        const float z = s + tb[0];
        delta[t] = (1.0f / (1.0f + __expf(-z))) * 0.99f + 0.01f;
    }
}

// 512 independent recurrences: thread = (batch, state)
__global__ __launch_bounds__(256)
void scan_kernel(const float* __restrict__ delta,
                 const __hip_bfloat16* __restrict__ Bv,
                 const float* __restrict__ Arow,
                 __hip_bfloat16* __restrict__ hs)
{
    const int id = blockIdx.x * 256 + threadIdx.x;
    if (id >= BATCH * DSTATE) return;
    const int b = id >> 4, n = id & 15;
    const float An = Arow[n];
    float h = 0.0f;
    for (int s = 0; s < SEQ; ++s) {
        const int t = b * SEQ + s;
        const float a = __expf(An * delta[t]);
        h = a * h + __bfloat162float(Bv[(size_t)t * DSTATE + n]);
        hs[(size_t)t * DSTATE + n] = __float2bfloat16(h);
    }
}

// hidden = hid * silu(gate) from packed h[M, 2*DHID]
__global__ __launch_bounds__(256)
void swiglu_kernel(const __hip_bfloat16* __restrict__ h,
                   __hip_bfloat16* __restrict__ out)
{
    const size_t idx = (size_t)blockIdx.x * 256 + threadIdx.x;
    if (idx >= (size_t)MTOK * DHID) return;
    const int m = (int)(idx / DHID);
    const int n = (int)(idx % DHID);
    const float hid = __bfloat162float(h[(size_t)m * (2 * DHID) + n]);
    const float gt  = __bfloat162float(h[(size_t)m * (2 * DHID) + DHID + n]);
    const float sg  = gt / (1.0f + __expf(-gt));
    out[idx] = __float2bfloat16(hid * sg);
}

// pooled[b, d] = mean over S tokens of xn (post final LN), stored bf16
__global__ __launch_bounds__(256)
void pool_kernel(const __hip_bfloat16* __restrict__ xn,
                 __hip_bfloat16* __restrict__ pooled)
{
    const int id = blockIdx.x * 256 + threadIdx.x;
    if (id >= BATCH * DMODEL) return;
    const int b = id / DMODEL, d = id % DMODEL;
    float s = 0.0f;
    for (int t = 0; t < SEQ; ++t)
        s += __bfloat162float(xn[((size_t)b * SEQ + t) * DMODEL + d]);
    pooled[id] = __float2bfloat16(s * (1.0f / SEQ));
}

// ---------------------------------------------------------------------------
// Host side
// ---------------------------------------------------------------------------
template<int MODE>
static void gemm(hipStream_t st, const __hip_bfloat16* A, int lda,
                 const float* W, const float* bias,
                 float* outF, __hip_bfloat16* outB,
                 const float* pos, const __hip_bfloat16* xn, const float* Dv,
                 int M, int N, int K)
{
    dim3 grid((N + 127) / 128, (M + 127) / 128);
    gemm_wmma<MODE><<<grid, 256, 0, st>>>(A, lda, W, bias, outF, outB, pos, xn, Dv, M, N, K);
}

static inline char* wsal(char*& p, size_t bytes) {
    char* r = p;
    p += (bytes + 255) & ~(size_t)255;
    return r;
}

extern "C" void kernel_launch(void* const* d_in, const int* in_sizes, int n_in,
                              void* d_out, int out_size, void* d_ws, size_t ws_size,
                              hipStream_t stream)
{
    (void)in_sizes; (void)n_in; (void)out_size; (void)ws_size;

    const float* imgs     = (const float*)d_in[0];
    const float* patch_W  = (const float*)d_in[1];
    const float* patch_b  = (const float*)d_in[2];
    const float* pos_emb  = (const float*)d_in[3];
    const float* n1g      = (const float*)d_in[4];
    const float* n1b      = (const float*)d_in[5];
    const float* n2g      = (const float*)d_in[6];
    const float* n2b      = (const float*)d_in[7];
    const float* Amat     = (const float*)d_in[8];
    const float* BW       = (const float*)d_in[9];
    const float* Bb       = (const float*)d_in[10];
    const float* CW       = (const float*)d_in[11];
    const float* Cb       = (const float*)d_in[12];
    const float* Dd       = (const float*)d_in[13];
    const float* tW       = (const float*)d_in[14];
    const float* tb       = (const float*)d_in[15];
    const float* oW       = (const float*)d_in[16];
    const float* ob       = (const float*)d_in[17];
    const float* miW      = (const float*)d_in[18];
    const float* mib      = (const float*)d_in[19];
    const float* moW      = (const float*)d_in[20];
    const float* mob      = (const float*)d_in[21];
    const float* nfg      = (const float*)d_in[22];
    const float* nfb      = (const float*)d_in[23];
    const float* cls_W    = (const float*)d_in[24];
    const float* cls_b    = (const float*)d_in[25];

    char* p = (char*)d_ws;
    __hip_bfloat16* patches = (__hip_bfloat16*)wsal(p, (size_t)MTOK * PDIM * 2);
    float*          x       = (float*)         wsal(p, (size_t)MTOK * DMODEL * 4);
    __hip_bfloat16* xn      = (__hip_bfloat16*)wsal(p, (size_t)MTOK * DMODEL * 2);
    __hip_bfloat16* ybuf    = (__hip_bfloat16*)wsal(p, (size_t)MTOK * DMODEL * 2);
    __hip_bfloat16* hbuf    = (__hip_bfloat16*)wsal(p, (size_t)MTOK * 2 * DHID * 2);
    __hip_bfloat16* hidden  = (__hip_bfloat16*)wsal(p, (size_t)MTOK * DHID * 2);
    float*          dbuf    = (float*)         wsal(p, (size_t)MTOK * 4);
    __hip_bfloat16* Bv      = (__hip_bfloat16*)wsal(p, (size_t)MTOK * DSTATE * 2);
    __hip_bfloat16* hs      = (__hip_bfloat16*)wsal(p, (size_t)MTOK * DSTATE * 2);
    __hip_bfloat16* pooled  = (__hip_bfloat16*)wsal(p, (size_t)BATCH * DMODEL * 2);

    // ---- patch embedding: x = patches @ patch_W^T + b + pos --------------
    {
        const size_t ne = (size_t)MTOK * PDIM;
        patchify_kernel<<<(unsigned)((ne + 255) / 256), 256, 0, stream>>>(imgs, patches);
        gemm<0>(stream, patches, PDIM, patch_W, patch_b, x, nullptr, pos_emb,
                nullptr, nullptr, MTOK, DMODEL, PDIM);
    }

    for (int i = 0; i < NLAYER; ++i) {
        // ---- SSM branch --------------------------------------------------
        ln_kernel<<<MTOK / 8, 256, 0, stream>>>(x, n1g + i * DMODEL, n1b + i * DMODEL, xn);
        delta_kernel<<<MTOK / 8, 256, 0, stream>>>(xn, tW + i * DMODEL, tb + i, dbuf);
        gemm<1>(stream, xn, DMODEL, BW + (size_t)i * DSTATE * DMODEL, Bb + i * DSTATE,
                nullptr, Bv, nullptr, nullptr, nullptr, MTOK, DSTATE, DMODEL);
        scan_kernel<<<2, 256, 0, stream>>>(dbuf, Bv, Amat + i * DSTATE, hs);
        gemm<2>(stream, hs, DSTATE, CW + (size_t)i * DMODEL * DSTATE, Cb + i * DMODEL,
                nullptr, ybuf, nullptr, xn, Dd + i * DMODEL, MTOK, DMODEL, DSTATE);
        gemm<3>(stream, ybuf, DMODEL, oW + (size_t)i * DMODEL * DMODEL, ob + i * DMODEL,
                x, nullptr, nullptr, nullptr, nullptr, MTOK, DMODEL, DMODEL);

        // ---- MLP branch --------------------------------------------------
        ln_kernel<<<MTOK / 8, 256, 0, stream>>>(x, n2g + i * DMODEL, n2b + i * DMODEL, xn);
        gemm<1>(stream, xn, DMODEL, miW + (size_t)i * 2 * DHID * DMODEL, mib + i * 2 * DHID,
                nullptr, hbuf, nullptr, nullptr, nullptr, MTOK, 2 * DHID, DMODEL);
        {
            const size_t ne = (size_t)MTOK * DHID;
            swiglu_kernel<<<(unsigned)((ne + 255) / 256), 256, 0, stream>>>(hbuf, hidden);
        }
        gemm<3>(stream, hidden, DHID, moW + (size_t)i * DMODEL * DHID, mob + i * DMODEL,
                x, nullptr, nullptr, nullptr, nullptr, MTOK, DMODEL, DHID);
    }

    // ---- head: final LN -> mean pool -> classifier -----------------------
    ln_kernel<<<MTOK / 8, 256, 0, stream>>>(x, nfg, nfb, xn);
    pool_kernel<<<(BATCH * DMODEL + 255) / 256, 256, 0, stream>>>(xn, pooled);
    gemm<0>(stream, pooled, DMODEL, cls_W, cls_b, (float*)d_out, nullptr, nullptr,
            nullptr, nullptr, BATCH, NCLS, DMODEL);
}
